// SwinTransformerBlock_36996848288291
// MI455X (gfx1250) — compile-verified, matrix operand order checked by reading, not run on hardware
//
#include <hip/hip_runtime.h>
#include <hip/hip_bf16.h>

// ---------------------------------------------------------------------------
// CDNA5 (gfx1250) fused Swin-3D block. wave32, WMMA bf16 16x16x32, f32 accum.
// - fragment loads: 2x16B contiguous per lane (ds_load_b128 / global_load_b128)
// - weights pre-swizzled into fragment order (coalesced b128, L2-resident)
// - wave->row-tile mapping: A fragments hoisted to registers, reused per nt
// - stage-0 window tile copy uses GLOBAL_LOAD_ASYNC_TO_LDS_B128 (+ ASYNCcnt)
// - rel-pos bias gather pre-folded into a dense [3][98][98] table
// ---------------------------------------------------------------------------

typedef __attribute__((ext_vector_type(16))) __bf16 v16bf;
typedef __attribute__((ext_vector_type(8)))  float  v8f;

namespace {
constexpr int Bb    = 8;
constexpr int Dd    = 16;
constexpr int Hh    = 56;
constexpr int Wwid  = 56;
constexpr int Cc    = 96;
constexpr int WD    = 2, WH = 7, WW = 7;
constexpr int SD    = 1, SH = 3, SW = 3;
constexpr int NH    = 3;
constexpr int HD    = 32;               // head dim
constexpr int Ntok  = 98;               // valid tokens per window
constexpr int NP    = 128;              // padded tokens (multiple of 32)
constexpr int NWimg = (Dd/WD)*(Hh/WH)*(Wwid/WW);  // 512
constexpr int NWIN  = Bb * NWimg;       // 4096
constexpr int HIDDEN = 4 * Cc;          // 384
constexpr float EPS   = 1e-5f;
constexpr float SCALE = 0.17677669529663687f;     // 1/sqrt(32)
constexpr int QKVC  = 3 * Cc;           // 288
}

// ---------------- helpers ----------------

__device__ __forceinline__ unsigned short f2bf(float f) {
  unsigned int u = __float_as_uint(f);
  u += 0x7FFFu + ((u >> 16) & 1u);      // round-to-nearest-even
  return (unsigned short)(u >> 16);
}

union BfFrag { v16bf v; unsigned short s[16]; uint4 q[2]; };

// A fragment: 16x32 bf16, src row-major [16][>=32], leading dim ld (halfs).
__device__ __forceinline__ v16bf load_a_frag(const unsigned short* src, int ld) {
  const int lane = threadIdx.x & 31;
  const unsigned short* p = src + (lane & 15) * ld + ((lane >> 4) << 3);
  BfFrag u;
  u.q[0] = *(const uint4*)(p);
  u.q[1] = *(const uint4*)(p + 16);
  return u.v;
}

// B fragment where logical B = srcT^T : srcT is [16][>=32] row-major (N-major).
__device__ __forceinline__ v16bf load_bt_frag(const unsigned short* srcT, int ld) {
  const int lane = threadIdx.x & 31;
  const unsigned short* p = srcT + (lane & 15) * ld + ((lane >> 4) << 3);
  BfFrag u;
  u.q[0] = *(const uint4*)(p);
  u.q[1] = *(const uint4*)(p + 16);
  return u.v;
}

// Pre-swizzled weight fragment: tile (kt,nt); each lane's 16 halfs contiguous.
__device__ __forceinline__ v16bf load_w_frag(const unsigned short* packed,
                                             int kt, int nt, int ktiles) {
  const int lane = threadIdx.x & 31;
  const uint4* p = (const uint4*)packed +
                   (((size_t)nt * ktiles + kt) * 32 + lane) * 2;
  BfFrag u;
  u.q[0] = p[0];
  u.q[1] = p[1];
  return u.v;
}

__device__ __forceinline__ v8f wmma_bf16(v16bf a, v16bf b, v8f c) {
  return __builtin_amdgcn_wmma_f32_16x16x32_bf16(false, a, false, b, (short)0, c,
                                                 false, false);
}

// Async copy 16B/lane from global to LDS (gfx1250 GLOBAL_LOAD_ASYNC_TO_LDS_B128,
// GV mode, tracked by ASYNCcnt).  lds_off = LDS byte offset, gaddr = 64-bit VA.
__device__ __forceinline__ void async_copy16(unsigned lds_off, const void* gaddr) {
  asm volatile("global_load_async_to_lds_b128 %0, %1, off"
               :: "v"(lds_off), "v"(gaddr)
               : "memory");
}
__device__ __forceinline__ void async_wait0() {
  asm volatile("s_wait_asynccnt 0" ::: "memory");
}

// Vectorized LayerNorm of one 96-float row -> 96 bf16 (packed uint2 stores).
__device__ __forceinline__ void layernorm_row(const float* __restrict__ src,
                                              const float* __restrict__ g,
                                              const float* __restrict__ bsh,
                                              unsigned short* __restrict__ dst) {
  const float4* s4 = (const float4*)src;
  float mu = 0.f;
#pragma unroll
  for (int j = 0; j < Cc / 4; ++j) {
    const float4 v = s4[j];
    mu += v.x + v.y + v.z + v.w;
  }
  mu *= (1.0f / Cc);
  float var = 0.f;
#pragma unroll
  for (int j = 0; j < Cc / 4; ++j) {
    const float4 v = s4[j];
    const float a = v.x - mu, b = v.y - mu, c = v.z - mu, d = v.w - mu;
    var += a * a + b * b + c * c + d * d;
  }
  var *= (1.0f / Cc);
  const float inv = rsqrtf(var + EPS);
  const float4* g4 = (const float4*)g;
  const float4* b4 = (const float4*)bsh;
  uint2* d2 = (uint2*)dst;
#pragma unroll
  for (int j = 0; j < Cc / 4; ++j) {
    const float4 v = s4[j];
    const float4 gg = g4[j];
    const float4 bb = b4[j];
    uint2 o;
    o.x = (unsigned)f2bf((v.x - mu) * inv * gg.x + bb.x) |
          ((unsigned)f2bf((v.y - mu) * inv * gg.y + bb.y) << 16);
    o.y = (unsigned)f2bf((v.z - mu) * inv * gg.z + bb.z) |
          ((unsigned)f2bf((v.w - mu) * inv * gg.w + bb.w) << 16);
    d2[j] = o;
  }
}

// ---------------- kernel 0: weight swizzle + bias-table fold ----------------

__device__ __forceinline__ void pack_weight(const float* __restrict__ w,
                                            unsigned short* __restrict__ o,
                                            int K, int Nn, int tid, int stride) {
  const int ktiles = K / 32;
  const int total = K * Nn;
  for (int oi = tid; oi < total; oi += stride) {
    const int e    = oi & 15;
    const int lane = (oi >> 4) & 31;
    const int tile = oi >> 9;                  // 512 halfs per 32x16 tile
    const int kt   = tile % ktiles;
    const int nt   = tile / ktiles;
    const int i = e >> 1, j = e & 1;
    const int k = kt * 32 + ((i < 4) ? 0 : 16) + ((lane >> 4) << 3) + ((i & 3) << 1) + j;
    const int n = nt * 16 + (lane & 15);
    o[oi] = f2bf(w[k * Nn + n]);
  }
}

__global__ void prep_tables(const float* __restrict__ wqkv, const float* __restrict__ wproj,
                            const float* __restrict__ wfc1, const float* __restrict__ wfc2,
                            const float* __restrict__ btab, const int* __restrict__ ridx,
                            unsigned short* __restrict__ o_qkv, unsigned short* __restrict__ o_proj,
                            unsigned short* __restrict__ o_fc1, unsigned short* __restrict__ o_fc2,
                            float* __restrict__ bias3) {
  const int tid = blockIdx.x * blockDim.x + threadIdx.x;
  const int stride = gridDim.x * blockDim.x;
  pack_weight(wqkv, o_qkv, Cc, QKVC, tid, stride);
  pack_weight(wproj, o_proj, Cc, Cc, tid, stride);
  pack_weight(wfc1, o_fc1, Cc, HIDDEN, tid, stride);
  pack_weight(wfc2, o_fc2, HIDDEN, Cc, tid, stride);
  // fold rel-pos gather: bias3[h][r][c] = bias_table[rel_index[r][c]][h]
  for (int i = tid; i < NH * Ntok * Ntok; i += stride) {
    const int h  = i / (Ntok * Ntok);
    const int rc = i - h * (Ntok * Ntok);
    bias3[i] = btab[ridx[rc] * NH + h];
  }
}

// ---------------- kernel 1: LN1 + cyclic shift + window partition -> bf16 ------

__global__ void ln1_partition(const float* __restrict__ x,
                              const float* __restrict__ g1, const float* __restrict__ b1,
                              unsigned short* __restrict__ xw) {
  const int gid = blockIdx.x * blockDim.x + threadIdx.x;
  if (gid >= NWIN * NP) return;
  const int win = gid / NP;
  const int tok = gid - win * NP;
  unsigned short* dst = xw + (size_t)gid * Cc;
  if (tok >= Ntok) {                      // zero padding rows
    uint2* d2 = (uint2*)dst;
#pragma unroll
    for (int j = 0; j < Cc / 4; ++j) d2[j] = make_uint2(0u, 0u);
    return;
  }
  const int b    = win / NWimg;
  const int widx = win - b * NWimg;
  const int wd = widx >> 6;
  const int wh = (widx >> 3) & 7;
  const int wb = widx & 7;
  const int dz  = tok / (WH * WW);
  const int rem = tok - dz * (WH * WW);
  const int hy  = rem / WW;
  const int wx  = rem - hy * WW;
  // shifted[p] = x[(p + S) % L]  (roll with shift=-S)
  const int ds = (wd * WD + dz + SD) % Dd;
  const int hs = (wh * WH + hy + SH) % Hh;
  const int wsc = (wb * WW + wx + SW) % Wwid;
  const float* src = x + ((((size_t)b * Dd + ds) * Hh + hs) * Wwid + wsc) * Cc;
  layernorm_row(src, g1, b1, dst);
}

// ---------------- kernel 2: fused windowed attention ----------------
// One block per window; 8 waves; wave w owns row-tile mt=w everywhere.

__global__ __launch_bounds__(256)
void attn_window(const unsigned short* __restrict__ xw,
                 const unsigned short* __restrict__ wqkv, const float* __restrict__ bqkv,
                 const unsigned short* __restrict__ wproj, const float* __restrict__ bproj,
                 const float* __restrict__ bias3,           // [3][98][98] (pre-folded)
                 const float* __restrict__ mask,            // [512][98][98]
                 const float* __restrict__ x0,              // residual source
                 float*       __restrict__ out) {
  extern __shared__ char smem[];
  unsigned short* sX   = (unsigned short*)smem;             // NP x 96   bf16
  unsigned short* sQKV = sX + NP * Cc;                      // NP x 288  bf16
  float*          sS   = (float*)(sQKV + NP * QKVC);        // NP x NP   f32
  unsigned short* sP   = (unsigned short*)(sS + NP * NP);   // NP x NP   bf16
  unsigned short* sO   = sP + NP * NP;                      // NP x 96   bf16
  unsigned short* sVt  = sO + NP * Cc;                      // 96 x NP   bf16 (V^T)

  const int win  = blockIdx.x;
  const int bidx = win / NWimg;
  const int widx = win - bidx * NWimg;
  const int tid  = threadIdx.x;
  const int wave = tid >> 5;              // == row-tile mt (8 waves, 8 tiles)
  const int lane = tid & 31;
  const int col  = lane & 15;
  const int rb   = (lane >> 4) << 3;
  const int mt   = wave;

  // --- stage 0: async copy of LN'd window tile straight into LDS ---
  {
    const char* gbase = (const char*)(xw + (size_t)win * NP * Cc);
    const unsigned lds_base = (unsigned)(size_t)(void*)sX;
#pragma unroll
    for (int it = 0; it < (NP * Cc * 2) / (256 * 16); ++it) {   // 6 iters
      const unsigned off = (unsigned)tid * 16u + (unsigned)it * 4096u;
      async_copy16(lds_base + off, gbase + off);
    }
    async_wait0();
  }
  __syncthreads();

  // --- stage 1: QKV = X @ Wqkv + b   ([NP x 288], K=96); A-frags hoisted ---
  {
    const v16bf a0 = load_a_frag(sX + mt * 16 * Cc + 0,  Cc);
    const v16bf a1 = load_a_frag(sX + mt * 16 * Cc + 32, Cc);
    const v16bf a2 = load_a_frag(sX + mt * 16 * Cc + 64, Cc);
    for (int nt = 0; nt < QKVC / 16; ++nt) {
      v8f acc = {};
      acc = wmma_bf16(a0, load_w_frag(wqkv, 0, nt, 3), acc);
      acc = wmma_bf16(a1, load_w_frag(wqkv, 1, nt, 3), acc);
      acc = wmma_bf16(a2, load_w_frag(wqkv, 2, nt, 3), acc);
      const int cb = nt * 16;
      const float bias = bqkv[cb + col];
      if (nt < 12) {                       // Q/K tiles (uniform branch)
#pragma unroll
        for (int i = 0; i < 8; ++i)
          sQKV[(mt * 16 + rb + i) * QKVC + cb + col] = f2bf(acc[i] + bias);
      } else {                             // V tiles: row copy + packed V^T copy
        unsigned short hv[8];
#pragma unroll
        for (int i = 0; i < 8; ++i) {
          hv[i] = f2bf(acc[i] + bias);
          sQKV[(mt * 16 + rb + i) * QKVC + cb + col] = hv[i];
        }
        uint4 pk;
        pk.x = (unsigned)hv[0] | ((unsigned)hv[1] << 16);
        pk.y = (unsigned)hv[2] | ((unsigned)hv[3] << 16);
        pk.z = (unsigned)hv[4] | ((unsigned)hv[5] << 16);
        pk.w = (unsigned)hv[6] | ((unsigned)hv[7] << 16);
        *(uint4*)(sVt + (cb + col - 2 * Cc) * NP + mt * 16 + rb) = pk;
      }
    }
  }
  __syncthreads();

  // --- per head: scores, softmax, P@V ---
  for (int h = 0; h < NH; ++h) {
    // S = scale * Q @ K^T + bias + mask  (K-dim = 32, one WMMA per tile).
    // Tile validity is uniform except the boundary tile index 6.
    {
      const v16bf aq = load_a_frag(sQKV + mt * 16 * QKVC + h * HD, QKVC);
      const float* bias_h = bias3 + (size_t)h * Ntok * Ntok;
      const float* mask_w = mask + (size_t)widx * Ntok * Ntok;
      for (int nt = 0; nt < NP / 16; ++nt) {
        const int c = nt * 16 + col;
        if (mt >= 7 || nt >= 7) {          // all-pad tile: no WMMA needed
#pragma unroll
          for (int i = 0; i < 8; ++i)
            sS[(mt * 16 + rb + i) * NP + c] = -1e30f;
          continue;
        }
        v8f acc = {};
        acc = wmma_bf16(aq, load_bt_frag(sQKV + nt * 16 * QKVC + Cc + h * HD, QKVC), acc);
        if (mt < 6 && nt < 6) {            // interior tile: branch-free
#pragma unroll
          for (int i = 0; i < 8; ++i) {
            const int r = mt * 16 + rb + i;
            sS[r * NP + c] = acc[i] * SCALE + bias_h[r * Ntok + c] + mask_w[r * Ntok + c];
          }
        } else {                           // boundary tile 6: per-element guard
          const bool cok = (c < Ntok);
#pragma unroll
          for (int i = 0; i < 8; ++i) {
            const int r = mt * 16 + rb + i;
            float v = -1e30f;
            if (r < Ntok && cok)
              v = acc[i] * SCALE + bias_h[r * Ntok + c] + mask_w[r * Ntok + c];
            sS[r * NP + c] = v;
          }
        }
      }
    }
    __syncthreads();

    // row softmax (one thread per row; b128 float4 passes, packed bf16 out)
    if (tid < NP) {
      float4* row4 = (float4*)(sS + tid * NP);
      float m = -1e30f;
#pragma unroll 4
      for (int j = 0; j < NP / 4; ++j) {
        const float4 v = row4[j];
        m = fmaxf(m, fmaxf(fmaxf(v.x, v.y), fmaxf(v.z, v.w)));
      }
      float s = 0.f;
#pragma unroll 4
      for (int j = 0; j < NP / 4; ++j) {
        float4 v = row4[j];
        v.x = __expf(v.x - m); v.y = __expf(v.y - m);
        v.z = __expf(v.z - m); v.w = __expf(v.w - m);
        s += v.x + v.y + v.z + v.w;
        row4[j] = v;
      }
      const float inv = 1.0f / s;
      uint2* prow = (uint2*)(sP + tid * NP);
#pragma unroll 4
      for (int j = 0; j < NP / 4; ++j) {
        const float4 v = row4[j];
        uint2 o;
        o.x = (unsigned)f2bf(v.x * inv) | ((unsigned)f2bf(v.y * inv) << 16);
        o.y = (unsigned)f2bf(v.z * inv) | ((unsigned)f2bf(v.w * inv) << 16);
        prow[j] = o;
      }
    }
    __syncthreads();

    // O_h = P @ V^T-copy   ([NP x 32], K = 128); A-frags hoisted across nt
    {
      v16bf ap[4];
#pragma unroll
      for (int k = 0; k < 4; ++k)
        ap[k] = load_a_frag(sP + mt * 16 * NP + k * 32, NP);
#pragma unroll
      for (int nt = 0; nt < HD / 16; ++nt) {
        v8f acc = {};
#pragma unroll
        for (int k = 0; k < 4; ++k) {
          const v16bf bv = load_bt_frag(sVt + (h * HD + nt * 16) * NP + k * 32, NP);
          acc = wmma_bf16(ap[k], bv, acc);
        }
#pragma unroll
        for (int i = 0; i < 8; ++i)
          sO[(mt * 16 + rb + i) * Cc + h * HD + nt * 16 + col] = f2bf(acc[i]);
      }
    }
    __syncthreads();
  }

  // --- proj + window reverse + inverse shift + residual -> out ---
  if (mt < 7) {                            // row-tile 7 is all padding
    const int wd = widx >> 6;
    const int wh = (widx >> 3) & 7;
    const int wb = widx & 7;
    const v16bf a0 = load_a_frag(sO + mt * 16 * Cc + 0,  Cc);
    const v16bf a1 = load_a_frag(sO + mt * 16 * Cc + 32, Cc);
    const v16bf a2 = load_a_frag(sO + mt * 16 * Cc + 64, Cc);
    for (int nt = 0; nt < Cc / 16; ++nt) {
      v8f acc = {};
      acc = wmma_bf16(a0, load_w_frag(wproj, 0, nt, 3), acc);
      acc = wmma_bf16(a1, load_w_frag(wproj, 1, nt, 3), acc);
      acc = wmma_bf16(a2, load_w_frag(wproj, 2, nt, 3), acc);
      const int c = nt * 16 + col;
      const float bias = bproj[c];
#pragma unroll
      for (int i = 0; i < 8; ++i) {
        const int r = mt * 16 + rb + i;
        if (mt < 6 || r < Ntok) {          // uniform for mt<6
          const int dz  = r / (WH * WW);
          const int rem = r - dz * (WH * WW);
          const int hy  = rem / WW;
          const int wx  = rem - hy * WW;
          // shifted position p -> final position (p + S) % L
          const int df = (wd * WD + dz + SD) % Dd;
          const int hf = (wh * WH + hy + SH) % Hh;
          const int wf = (wb * WW + wx + SW) % Wwid;
          const size_t idx = ((((size_t)bidx * Dd + df) * Hh + hf) * Wwid + wf) * Cc + c;
          out[idx] = x0[idx] + acc[i] + bias;
        }
      }
    }
  }
}

// ---------------- kernel 3: fused LN2 + fc1 + GELU + fc2 + residual ------------
// 64-token tile per block; fc1 activation lives only in LDS.

__global__ __launch_bounds__(256)
void mlp_block(float* __restrict__ xio,
               const float* __restrict__ g2, const float* __restrict__ b2,
               const unsigned short* __restrict__ wfc1, const float* __restrict__ bfc1,
               const unsigned short* __restrict__ wfc2, const float* __restrict__ bfc2) {
  __shared__ unsigned short sH[64 * Cc];       // 12 KB
  __shared__ unsigned short sA[64 * HIDDEN];   // 48 KB
  const int t0   = blockIdx.x * 64;
  const int tid  = threadIdx.x;
  const int wave = tid >> 5;
  const int lane = tid & 31;
  const int col  = lane & 15;
  const int rb   = (lane >> 4) << 3;
  const int mt   = wave >> 1;                  // 4 row-tiles, 2 waves each

  if (tid < 64)
    layernorm_row(xio + (size_t)(t0 + tid) * Cc, g2, b2, sH + tid * Cc);
  __syncthreads();

  // fc1 + exact GELU  ([64 x 384], K=96); wave covers 12 nt with hoisted A
  {
    const int ntb = (wave & 1) * 12;
    const v16bf a0 = load_a_frag(sH + mt * 16 * Cc + 0,  Cc);
    const v16bf a1 = load_a_frag(sH + mt * 16 * Cc + 32, Cc);
    const v16bf a2 = load_a_frag(sH + mt * 16 * Cc + 64, Cc);
    for (int ntoff = 0; ntoff < 12; ++ntoff) {
      const int nt = ntb + ntoff;
      v8f acc = {};
      acc = wmma_bf16(a0, load_w_frag(wfc1, 0, nt, 3), acc);
      acc = wmma_bf16(a1, load_w_frag(wfc1, 1, nt, 3), acc);
      acc = wmma_bf16(a2, load_w_frag(wfc1, 2, nt, 3), acc);
      const int c = nt * 16 + col;
      const float bias = bfc1[c];
#pragma unroll
      for (int i = 0; i < 8; ++i) {
        float v = acc[i] + bias;
        v = 0.5f * v * (1.0f + erff(v * 0.70710678118654752f));
        sA[(mt * 16 + rb + i) * HIDDEN + c] = f2bf(v);
      }
    }
  }
  __syncthreads();

  // fc2 + residual  ([64 x 96], K=384); 3 live accumulators across K loop
  {
    const int ntb = (wave & 1) * 3;
    v8f acc0 = {}, acc1 = {}, acc2 = {};
    for (int kt = 0; kt < HIDDEN / 32; ++kt) {
      const v16bf a = load_a_frag(sA + mt * 16 * HIDDEN + kt * 32, HIDDEN);
      acc0 = wmma_bf16(a, load_w_frag(wfc2, kt, ntb + 0, HIDDEN / 32), acc0);
      acc1 = wmma_bf16(a, load_w_frag(wfc2, kt, ntb + 1, HIDDEN / 32), acc1);
      acc2 = wmma_bf16(a, load_w_frag(wfc2, kt, ntb + 2, HIDDEN / 32), acc2);
    }
    v8f accs[3] = {acc0, acc1, acc2};
#pragma unroll
    for (int j = 0; j < 3; ++j) {
      const int c = (ntb + j) * 16 + col;
      const float bias = bfc2[c];
#pragma unroll
      for (int i = 0; i < 8; ++i) {
        const size_t idx = (size_t)(t0 + mt * 16 + rb + i) * Cc + c;
        xio[idx] = xio[idx] + accs[j][i] + bias;   // element owned by one lane
      }
    }
  }
}

// ---------------- host launch ----------------

extern "C" void kernel_launch(void* const* d_in, const int* in_sizes, int n_in,
                              void* d_out, int out_size, void* d_ws, size_t ws_size,
                              hipStream_t stream) {
  (void)in_sizes; (void)n_in; (void)out_size; (void)ws_size;
  const float* x      = (const float*)d_in[0];
  const float* mask   = (const float*)d_in[1];
  const float* g1     = (const float*)d_in[2];
  const float* b1     = (const float*)d_in[3];
  const float* w_qkv  = (const float*)d_in[4];
  const float* b_qkv  = (const float*)d_in[5];
  const float* w_proj = (const float*)d_in[6];
  const float* b_proj = (const float*)d_in[7];
  const float* btab   = (const float*)d_in[8];
  const int*   ridx   = (const int*)d_in[9];
  const float* g2     = (const float*)d_in[10];
  const float* b2     = (const float*)d_in[11];
  const float* w_fc1  = (const float*)d_in[12];
  const float* b_fc1  = (const float*)d_in[13];
  const float* w_fc2  = (const float*)d_in[14];
  const float* b_fc2  = (const float*)d_in[15];
  float* out = (float*)d_out;

  // workspace: xw bf16 [4096*128*96] (~96MB), swizzled weights, folded bias.
  unsigned short* ws     = (unsigned short*)d_ws;
  unsigned short* xw     = ws;
  unsigned short* qkv_b  = xw + (size_t)NWIN * NP * Cc;
  unsigned short* proj_b = qkv_b + Cc * QKVC;
  unsigned short* fc1_b  = proj_b + Cc * Cc;
  unsigned short* fc2_b  = fc1_b + Cc * HIDDEN;
  float*          bias3  = (float*)(fc2_b + HIDDEN * Cc);   // [3][98][98]

  prep_tables<<<96, 256, 0, stream>>>(w_qkv, w_proj, w_fc1, w_fc2, btab, ridx,
                                      qkv_b, proj_b, fc1_b, fc2_b, bias3);

  {
    const int total = NWIN * NP;
    ln1_partition<<<(total + 255) / 256, 256, 0, stream>>>(x, g1, b1, xw);
  }

  const int smem_bytes = NP * Cc * 2      // sX
                       + NP * QKVC * 2    // sQKV
                       + NP * NP * 4      // sS
                       + NP * NP * 2      // sP
                       + NP * Cc * 2      // sO
                       + Cc * NP * 2;     // sVt  = 245760 B (<320KB WGP LDS)
  (void)hipFuncSetAttribute(reinterpret_cast<const void*>(attn_window),
                            hipFuncAttributeMaxDynamicSharedMemorySize, smem_bytes);
  attn_window<<<NWIN, 256, smem_bytes, stream>>>(xw, qkv_b, b_qkv, proj_b, b_proj,
                                                 bias3, mask, x, out);

  mlp_block<<<(Bb * Dd * Hh * Wwid) / 64, 256, 0, stream>>>(out, g2, b2,
                                                            fc1_b, b_fc1, fc2_b, b_fc2);
}
